// HierarchicalMambaBlock_87290915324132
// MI455X (gfx1250) — matrix operand order repaired and checked by reading, not compiled, not run on hardware
//
#include <hip/hip_runtime.h>

// ---------------------------------------------------------------------------
// HierarchicalMambaBlock for MI455X (gfx1250), wave32 + WMMA bf16 matrix cores
// B=2, T=1024, DIM=512, D_INNER=1024, DT_RANK=64, D_STATE=16, CONV_K=4, NS=3
// ---------------------------------------------------------------------------

#define Bc 2
#define Tc 1024
#define DIMc 512
#define DIc 1024   // D_INNER
#define DTRANKc 64
#define DSTATEc 16
#define NPROJ 96   // DT_RANK + 2*D_STATE

typedef __attribute__((ext_vector_type(16))) __bf16 v16bf;
typedef __attribute__((ext_vector_type(4)))  __bf16 v4bf;
typedef __attribute__((ext_vector_type(8)))  float  v8f;

__device__ __forceinline__ float sigmoidf_(float x) { return 1.0f / (1.0f + __expf(-x)); }
__device__ __forceinline__ float siluf_(float x)    { return x * sigmoidf_(x); }
__device__ __forceinline__ float softplusf_(float x){ return (x > 20.0f) ? x : log1pf(__expf(x)); }

// ---------------------------------------------------------------------------
// Generic GEMM: D = act(A(MxK) * W(NxK)^T + bias + resid), f32 in/out,
// bf16 WMMA 16x16x32, f32 accumulate.
// Block tile 128x64, BK=32, 256 threads = 8 waves (4x2); wave tile 32x32
// (2 A-frags x 2 B-frags = 4 WMMAs / K-step). bf16 tiles staged in LDS,
// double-buffered (one barrier per K-step). Requires K % 32 == 0.
// ---------------------------------------------------------------------------
#define BM 128
#define BN 64
#define BK 32

__global__ __launch_bounds__(256) void gemm_wmma_bf16(
    const float* __restrict__ A, int lda,
    const float* __restrict__ W, int ldw,
    float* __restrict__ D, int ldd,
    const float* __restrict__ bias,
    const float* __restrict__ resid, int ldr,
    int M, int N, int K, int act)
{
    __shared__ __bf16 sA[2][BM][BK];   // 2 * 8KB
    __shared__ __bf16 sW[2][BN][BK];   // 2 * 4KB

    const int tid  = threadIdx.x;
    const int lane = tid & 31;
    const int wave = tid >> 5;
    const int lo   = lane & 15;
    const int hi   = lane >> 4;
    const int wm   = (wave >> 1) * 32;   // 0,32,64,96
    const int wn   = (wave & 1) * 32;    // 0,32
    const int blockM = blockIdx.y * BM;
    const int blockN = blockIdx.x * BN;

    v8f acc[2][2] = { { {}, {} }, { {}, {} } };

    float4 ra[4];   // staged A slab (4 float4 / thread = 128x32)
    float4 rw[2];   // staged W slab (2 float4 / thread = 64x32)

    const int nK = K / BK;

    // ---- cooperative global load of K-slab k0 into registers ----
    auto load_slab = [&](int k0) {
#pragma unroll
        for (int it = 0; it < 4; ++it) {
            int q   = tid + it * 256;        // 0..1023
            int row = q >> 3;                // 0..127
            int col = (q & 7) << 2;          // 0..28
            float4 v = make_float4(0.f, 0.f, 0.f, 0.f);
            int gm = blockM + row;
            if (gm < M) {
                const float* p = A + (size_t)gm * lda + k0 + col;
                v = *reinterpret_cast<const float4*>(p);
                if (k0 + 2 * BK < K) __builtin_prefetch(p + 2 * BK, 0, 1);
            }
            ra[it] = v;
        }
#pragma unroll
        for (int it = 0; it < 2; ++it) {
            int q   = tid + it * 256;        // 0..511
            int row = q >> 3;                // 0..63
            int col = (q & 7) << 2;
            float4 v = make_float4(0.f, 0.f, 0.f, 0.f);
            int gn = blockN + row;
            if (gn < N) {
                const float* p = W + (size_t)gn * ldw + k0 + col;
                v = *reinterpret_cast<const float4*>(p);
                if (k0 + 2 * BK < K) __builtin_prefetch(p + 2 * BK, 0, 1);
            }
            rw[it] = v;
        }
    };

    // ---- convert f32 -> bf16 once, store packed into LDS buffer ----
    auto store_slab = [&](int buf) {
#pragma unroll
        for (int it = 0; it < 4; ++it) {
            int q   = tid + it * 256;
            int row = q >> 3;
            int col = (q & 7) << 2;
            v4bf p;
            p.x = (__bf16)ra[it].x; p.y = (__bf16)ra[it].y;
            p.z = (__bf16)ra[it].z; p.w = (__bf16)ra[it].w;
            *reinterpret_cast<v4bf*>(&sA[buf][row][col]) = p;
        }
#pragma unroll
        for (int it = 0; it < 2; ++it) {
            int q   = tid + it * 256;
            int row = q >> 3;
            int col = (q & 7) << 2;
            v4bf p;
            p.x = (__bf16)rw[it].x; p.y = (__bf16)rw[it].y;
            p.z = (__bf16)rw[it].z; p.w = (__bf16)rw[it].w;
            *reinterpret_cast<v4bf*>(&sW[buf][row][col]) = p;
        }
    };

    load_slab(0);
    store_slab(0);
    __syncthreads();

    for (int kb = 0; kb < nK; ++kb) {
        const int cur  = kb & 1;
        const bool more = (kb + 1 < nK);
        if (more) load_slab((kb + 1) * BK);

        // Build fragments from LDS (ISA 7.12.2 wave32 layouts) and run 4 WMMAs.
        v16bf afr[2], bfr[2];
#pragma unroll
        for (int i = 0; i < 16; ++i) {
            int j = i >> 1, h = i & 1;
            int kkA = ((j & 4) ? 16 : 0) + hi * 8 + (j & 3) * 2 + h;
            int kkB = ((j & 4) ? 16 : 0) + hi * 8 + (j & 3) * 2 + h;
            afr[0][i] = sA[cur][wm + lo][kkA];
            afr[1][i] = sA[cur][wm + 16 + lo][kkA];
            bfr[0][i] = sW[cur][wn + lo][kkB];
            bfr[1][i] = sW[cur][wn + 16 + lo][kkB];
        }
#pragma unroll
        for (int am = 0; am < 2; ++am)
#pragma unroll
            for (int bn = 0; bn < 2; ++bn)
                acc[am][bn] = __builtin_amdgcn_wmma_f32_16x16x32_bf16(
                    false, afr[am], false, bfr[bn], (short)0, acc[am][bn],
                    false, false);

        if (more) store_slab(cur ^ 1);
        __syncthreads();
    }

    // Epilogue. C/D layout: VGPR r -> row (hi*8 + r), col = lo.
#pragma unroll
    for (int bn = 0; bn < 2; ++bn) {
        int gn = blockN + wn + bn * 16 + lo;
        float bv = (bias != nullptr && gn < N) ? bias[gn] : 0.0f;
#pragma unroll
        for (int am = 0; am < 2; ++am) {
#pragma unroll
            for (int r = 0; r < 8; ++r) {
                int gm = blockM + wm + am * 16 + hi * 8 + r;
                if (gm < M && gn < N) {
                    float v = acc[am][bn][r] + bv;
                    if (resid) v += resid[(size_t)gm * ldr + gn];
                    if (act == 1)      v = siluf_(v);
                    else if (act == 2) v = sigmoidf_(v);
                    D[(size_t)gm * ldd + gn] = v;
                }
            }
        }
    }
}

// ---------------------------------------------------------------------------
// Mean-pool downsample along time (stride 2 or 4), src may be strided (xz).
// ---------------------------------------------------------------------------
__global__ void downsample_kernel(const float* __restrict__ src, int ld, int coff,
                                  float* __restrict__ dst,
                                  int Bb, int Tout, int stride, int Dd)
{
    int gid = blockIdx.x * blockDim.x + threadIdx.x;
    int n = Bb * Tout * Dd;
    if (gid >= n) return;
    int d = gid % Dd;
    int t = (gid / Dd) % Tout;
    int b = gid / (Dd * Tout);
    int Tin = Tout * stride;
    float s = 0.f;
    for (int j = 0; j < stride; ++j)
        s += src[((size_t)(b * Tin + t * stride + j)) * ld + coff + d];
    dst[gid] = s * (1.0f / (float)stride);
}

// ---------------------------------------------------------------------------
// Depthwise causal conv1d (K=4) + bias + SiLU.  src strided, dst packed.
// ---------------------------------------------------------------------------
__global__ void dwconv_silu_kernel(const float* __restrict__ src, int ld, int coff,
                                   const float* __restrict__ w,
                                   const float* __restrict__ bias,
                                   float* __restrict__ dst, int Bb, int Tt, int Dd)
{
    int gid = blockIdx.x * blockDim.x + threadIdx.x;
    int n = Bb * Tt * Dd;
    if (gid >= n) return;
    int d = gid % Dd;
    int t = (gid / Dd) % Tt;
    int b = gid / (Dd * Tt);
    float acc = bias[d];
#pragma unroll
    for (int k = 0; k < 4; ++k) {
        int ts = t - 3 + k;
        if (ts >= 0)
            acc += w[d * 4 + k] * src[((size_t)(b * Tt + ts)) * ld + coff + d];
    }
    dst[gid] = siluf_(acc);
}

// ---------------------------------------------------------------------------
// Selective scan. One lane per (b, d, state s): 16-lane groups share a chain.
// h_t = max(exp(softplus(dt)*A_s),1e-38)*h_{t-1} + max(softplus(dt)*B_t*x_t,1e-38)
// y_t = sum_s C_t,s * h_t,s + D_d * x_t      (16-lane shfl-xor reduction)
// ---------------------------------------------------------------------------
__global__ __launch_bounds__(256) void scan_kernel(
    const float* __restrict__ xc, const float* __restrict__ dtb,
    const float* __restrict__ proj, const float* __restrict__ A_log,
    const float* __restrict__ Dp, float* __restrict__ y,
    int Bb, int Tt, int Dd)
{
    int gid = blockIdx.x * blockDim.x + threadIdx.x;
    int chain = gid >> 4;
    int s = gid & 15;
    if (chain >= Bb * Dd) return;
    int b = chain / Dd;
    int d = chain % Dd;

    const float A_s  = -__expf(A_log[d * DSTATEc + s]);
    const float Dp_d = Dp[d];

    float h = 0.0f;
    for (int t = 0; t < Tt; ++t) {
        size_t row = (size_t)(b * Tt + t);
        float xv  = xc[row * Dd + d];          // broadcast within 16-group
        float dtr = dtb[row * Dd + d];
        float Bv  = proj[row * NPROJ + DTRANKc + s];
        float Cv  = proj[row * NPROJ + DTRANKc + DSTATEc + s];

        float dtv = softplusf_(dtr);
        float dA  = fmaxf(__expf(dtv * A_s), 1e-38f);
        float dBx = fmaxf(dtv * Bv * xv, 1e-38f);
        h = fmaf(dA, h, dBx);

        float part = Cv * h;
        part += __shfl_xor(part, 8, 16);
        part += __shfl_xor(part, 4, 16);
        part += __shfl_xor(part, 2, 16);
        part += __shfl_xor(part, 1, 16);
        if (s == 0)
            y[row * Dd + d] = part + Dp_d * xv;
    }
}

// ---------------------------------------------------------------------------
// Upsample (nearest repeat) + softmax-weighted fuse + ctx mean.
// ---------------------------------------------------------------------------
__global__ void fuse_kernel(const float* __restrict__ y0, const float* __restrict__ y1,
                            const float* __restrict__ y2, const float* __restrict__ sw,
                            float* __restrict__ fused, float* __restrict__ ctx,
                            int Bb, int Tt, int Dd)
{
    int gid = blockIdx.x * blockDim.x + threadIdx.x;
    int n = Bb * Tt * Dd;
    if (gid >= n) return;
    int d = gid % Dd;
    int t = (gid / Dd) % Tt;
    int b = gid / (Dd * Tt);

    float v0 = y0[gid];
    float v1 = y1[((size_t)(b * (Tt / 2) + (t >> 1))) * Dd + d];
    float v2 = y2[((size_t)(b * (Tt / 4) + (t >> 2))) * Dd + d];

    float s0 = sw[0], s1 = sw[1], s2 = sw[2];
    float m  = fmaxf(s0, fmaxf(s1, s2));
    float e0 = __expf(s0 - m), e1 = __expf(s1 - m), e2 = __expf(s2 - m);
    float inv = 1.0f / (e0 + e1 + e2);

    fused[gid] = (e0 * v0 + e1 * v1 + e2 * v2) * inv;
    ctx[gid]   = (v0 + v1 + v2) * (1.0f / 3.0f);
}

// fused *= cg * silu(gate);  gate lives in xz[..., DI : 2*DI]
__global__ void gatemul_kernel(float* __restrict__ fused, const float* __restrict__ cg,
                               const float* __restrict__ xz, int nrows, int Dd,
                               int ldxz, int goff)
{
    int gid = blockIdx.x * blockDim.x + threadIdx.x;
    if (gid >= nrows * Dd) return;
    int row = gid / Dd;
    int d   = gid % Dd;
    float g = xz[(size_t)row * ldxz + goff + d];
    fused[gid] *= cg[gid] * siluf_(g);
}

// Row LayerNorm over Dd=512, one block per row.
__global__ __launch_bounds__(256) void layernorm_kernel(
    const float* __restrict__ y, const float* __restrict__ gamma,
    const float* __restrict__ beta, float* __restrict__ out, int Dd)
{
    __shared__ float ssum[256], ssq[256];
    int row = blockIdx.x;
    int tid = threadIdx.x;
    float s = 0.f, q = 0.f;
    for (int i = tid; i < Dd; i += 256) {
        float v = y[(size_t)row * Dd + i];
        s += v; q += v * v;
    }
    ssum[tid] = s; ssq[tid] = q;
    __syncthreads();
    for (int off = 128; off > 0; off >>= 1) {
        if (tid < off) { ssum[tid] += ssum[tid + off]; ssq[tid] += ssq[tid + off]; }
        __syncthreads();
    }
    float mu  = ssum[0] / (float)Dd;
    float var = ssq[0] / (float)Dd - mu * mu;
    float inv = rsqrtf(var + 1e-5f);
    for (int i = tid; i < Dd; i += 256) {
        float v = y[(size_t)row * Dd + i];
        out[(size_t)row * Dd + i] = (v - mu) * inv * gamma[i] + beta[i];
    }
}

// ---------------------------------------------------------------------------
extern "C" void kernel_launch(void* const* d_in, const int* in_sizes, int n_in,
                              void* d_out, int out_size, void* d_ws, size_t ws_size,
                              hipStream_t stream)
{
    (void)in_sizes; (void)n_in; (void)out_size; (void)ws_size;

    const float* x          = (const float*)d_in[0];
    const float* in_proj_w  = (const float*)d_in[1];
    const float* conv_w     = (const float*)d_in[2];
    const float* conv_b     = (const float*)d_in[3];
    const float* xproj_w    = (const float*)d_in[4];
    const float* dtproj_w   = (const float*)d_in[5];
    const float* dtproj_b   = (const float*)d_in[6];
    const float* A_log      = (const float*)d_in[7];
    const float* D_p        = (const float*)d_in[8];
    const float* scale_w    = (const float*)d_in[9];
    const float* cg_w1      = (const float*)d_in[10];
    const float* cg_w2      = (const float*)d_in[11];
    const float* out_proj_w = (const float*)d_in[12];
    const float* ln_gamma   = (const float*)d_in[13];
    const float* ln_beta    = (const float*)d_in[14];
    float* out = (float*)d_out;

    const int Ts[3] = { Tc, Tc / 2, Tc / 4 };

    // ---- workspace carve-up (f32, 256B aligned slabs) ----
    float* ws = (float*)d_ws;
    size_t off = 0;
    auto alloc = [&](size_t nelem) {
        float* p = ws + off;
        off += (nelem + 63) & ~(size_t)63;
        return p;
    };
    float* xz   = alloc((size_t)Bc * Tc * 2 * DIc);       // (B,T,2048)
    float* xs1  = alloc((size_t)Bc * Ts[1] * DIc);
    float* xs2  = alloc((size_t)Bc * Ts[2] * DIc);
    float* xcb[3]; float* projb[3]; float* dtb[3]; float* yb[3];
    for (int s = 0; s < 3; ++s) xcb[s]   = alloc((size_t)Bc * Ts[s] * DIc);
    for (int s = 0; s < 3; ++s) projb[s] = alloc((size_t)Bc * Ts[s] * NPROJ);
    for (int s = 0; s < 3; ++s) dtb[s]   = alloc((size_t)Bc * Ts[s] * DIc);
    for (int s = 0; s < 3; ++s) yb[s]    = alloc((size_t)Bc * Ts[s] * DIc);
    float* fused = alloc((size_t)Bc * Tc * DIc);
    float* ctx   = alloc((size_t)Bc * Tc * DIc);
    float* tmp1  = alloc((size_t)Bc * Tc * (DIc / 2));
    float* cgb   = alloc((size_t)Bc * Tc * DIc);
    float* ybuf  = alloc((size_t)Bc * Tc * DIMc);

    auto gemm = [&](const float* A, int lda, const float* W, int ldw,
                    float* Dst, int ldd, const float* bias,
                    const float* resid, int ldr, int M, int N, int K, int act) {
        dim3 grid((N + BN - 1) / BN, (M + BM - 1) / BM);
        gemm_wmma_bf16<<<grid, dim3(256), 0, stream>>>(A, lda, W, ldw, Dst, ldd,
                                                       bias, resid, ldr, M, N, K, act);
    };

    const int MT = Bc * Tc;   // 2048 tokens

    // 1) xz = x @ in_proj_w^T   (2048 x 2048, K=512)
    gemm(x, DIMc, in_proj_w, DIMc, xz, 2 * DIc, nullptr, nullptr, 0,
         MT, 2 * DIc, DIMc, 0);

    // 2) downsample x_in (xz[...,:DI]) for scales 1,2
    {
        int n1 = Bc * Ts[1] * DIc;
        downsample_kernel<<<(n1 + 255) / 256, 256, 0, stream>>>(xz, 2 * DIc, 0, xs1, Bc, Ts[1], 2, DIc);
        int n2 = Bc * Ts[2] * DIc;
        downsample_kernel<<<(n2 + 255) / 256, 256, 0, stream>>>(xz, 2 * DIc, 0, xs2, Bc, Ts[2], 4, DIc);
    }

    // 3) per-scale SSM core
    for (int s = 0; s < 3; ++s) {
        const float* src  = (s == 0) ? xz  : (s == 1 ? xs1 : xs2);
        int ld            = (s == 0) ? 2 * DIc : DIc;
        int Msc           = Bc * Ts[s];

        int nc = Msc * DIc;
        dwconv_silu_kernel<<<(nc + 255) / 256, 256, 0, stream>>>(
            src, ld, 0, conv_w + (size_t)s * DIc * 4, conv_b + (size_t)s * DIc,
            xcb[s], Bc, Ts[s], DIc);

        // proj = xc @ xproj_w[s]^T   (Msc x 96, K=1024)
        gemm(xcb[s], DIc, xproj_w + (size_t)s * NPROJ * DIc, DIc,
             projb[s], NPROJ, nullptr, nullptr, 0, Msc, NPROJ, DIc, 0);

        // dt = proj[:, :64] @ dtproj_w[s]^T + dtproj_b[s]   (Msc x 1024, K=64)
        gemm(projb[s], NPROJ, dtproj_w + (size_t)s * DIc * DTRANKc, DTRANKc,
             dtb[s], DIc, dtproj_b + (size_t)s * DIc, nullptr, 0,
             Msc, DIc, DTRANKc, 0);

        // selective scan
        int nthreads = Bc * DIc * DSTATEc;   // 32768
        scan_kernel<<<nthreads / 256, 256, 0, stream>>>(
            xcb[s], dtb[s], projb[s], A_log + (size_t)s * DIc * DSTATEc,
            D_p + (size_t)s * DIc, yb[s], Bc, Ts[s], DIc);
    }

    // 4) upsample + softmax fuse + ctx mean
    {
        int n = Bc * Tc * DIc;
        fuse_kernel<<<(n + 255) / 256, 256, 0, stream>>>(
            yb[0], yb[1], yb[2], scale_w, fused, ctx, Bc, Tc, DIc);
    }

    // 5) context gate: silu(ctx @ cg_w1^T) @ cg_w2^T, sigmoid
    gemm(ctx, DIc, cg_w1, DIc, tmp1, DIc / 2, nullptr, nullptr, 0,
         MT, DIc / 2, DIc, 1 /*silu*/);
    gemm(tmp1, DIc / 2, cg_w2, DIc / 2, cgb, DIc, nullptr, nullptr, 0,
         MT, DIc, DIc / 2, 2 /*sigmoid*/);

    // 6) fused *= cg * silu(gate)
    {
        int n = MT * DIc;
        gatemul_kernel<<<(n + 255) / 256, 256, 0, stream>>>(
            fused, cgb, xz, MT, DIc, 2 * DIc, DIc);
    }

    // 7) out = fused @ out_proj_w^T + residual(x)
    gemm(fused, DIc, out_proj_w, DIc, ybuf, DIMc, nullptr, x, DIMc,
         MT, DIMc, DIc, 0);

    // 8) layernorm -> d_out
    layernorm_kernel<<<MT, 256, 0, stream>>>(ybuf, ln_gamma, ln_beta, out, DIMc);
}
